// SelfAttention_11450382811738
// MI455X (gfx1250) — matrix operand order repaired and checked by reading, not compiled, not run on hardware
//
#include <hip/hip_runtime.h>
#include <hip/hip_bf16.h>
#include <cstdint>

#define EMBED 1024
#define HEADS 16
#define HDIM  64
#define BATCH 4
#define SEQ   2048

typedef __attribute__((ext_vector_type(16))) __bf16 v16bf;
typedef __attribute__((ext_vector_type(8)))  __bf16 v8bf;
typedef __attribute__((ext_vector_type(8)))  float  v8f;

static __device__ __forceinline__ v8f wmma_bf16(v16bf a, v16bf b, v8f c) {
  return __builtin_amdgcn_wmma_f32_16x16x32_bf16(false, a, false, b, (short)0, c, false, false);
}

// A fragment (16x32, M x K), row-major source, `stride` elements between rows.
// Lane: m = lane&15, hgrp = lane>>4. elems 0..7 -> K = hgrp*8+e ; elems 8..15 -> K = 16+hgrp*8+(e-8).
static __device__ __forceinline__ v16bf load_a_bf16(const __bf16* base, int stride) {
  const int lane = threadIdx.x & 31;
  const int m = lane & 15, hgrp = lane >> 4;
  const __bf16* p0 = base + m * stride + hgrp * 8;
  union { v16bf v; v8bf h[2]; } u;
  u.h[0] = *(const v8bf*)p0;
  u.h[1] = *(const v8bf*)(p0 + 16);
  return u.v;
}

static __device__ __forceinline__ v16bf load_a_f32(const float* base, int stride) {
  const int lane = threadIdx.x & 31;
  const int m = lane & 15, hgrp = lane >> 4;
  const float* p0 = base + m * stride + hgrp * 8;
  const float* p1 = p0 + 16;
  v16bf a;
#pragma unroll
  for (int e = 0; e < 8; ++e) { a[e] = (__bf16)p0[e]; a[8 + e] = (__bf16)p1[e]; }
  return a;
}

// B fragment (32x16, K x N): column n contiguous along K, colstride elems between columns.
// Lane: n = lane&15, khalf = lane>>4. elems 0..7 -> K = khalf*8+e ; elems 8..15 -> K = 16+khalf*8+(e-8).
static __device__ __forceinline__ v16bf load_b_bf16(const __bf16* base, int colstride) {
  const int lane = threadIdx.x & 31;
  const int n = lane & 15, khalf = lane >> 4;
  const __bf16* p = base + n * colstride + khalf * 8;
  union { v16bf v; v8bf h[2]; } u;
  u.h[0] = *(const v8bf*)p;
  u.h[1] = *(const v8bf*)(p + 16);
  return u.v;
}

// Per-lane async 16B copy global -> LDS (tracked by ASYNCcnt).
static __device__ __forceinline__ void async_b128(unsigned lds_off, const void* gptr) {
  asm volatile("global_load_async_to_lds_b128 %0, %1, off"
               :: "v"(lds_off), "v"(gptr) : "memory");
}

// ---------------- Kernel 1: weight conversion f32 -> bf16 ----------------
__global__ void cvt_weights(const float* __restrict__ Wv, const float* __restrict__ Wk,
                            const float* __restrict__ Wq, const float* __restrict__ Wo,
                            __bf16* __restrict__ Wvb, __bf16* __restrict__ Wkb,
                            __bf16* __restrict__ Wqb, __bf16* __restrict__ Wob) {
  int i = blockIdx.x * blockDim.x + threadIdx.x;
  if (i < HDIM * HDIM) {
    Wvb[i] = (__bf16)Wv[i];
    Wkb[i] = (__bf16)Wk[i];
    Wqb[i] = (__bf16)Wq[i];
  }
  if (i < EMBED * EMBED) Wob[i] = (__bf16)Wo[i];
}

// ---------------- Kernel 2: per-head QKV projection (WMMA) ----------------
__global__ void __launch_bounds__(32) proj_qkv(
    const float* __restrict__ query, const float* __restrict__ keys, const float* __restrict__ vals,
    const __bf16* __restrict__ Wqb, const __bf16* __restrict__ Wkb, const __bf16* __restrict__ Wvb,
    __bf16* __restrict__ Qp, __bf16* __restrict__ Kp, __bf16* __restrict__ Vt) {
  const int LT = SEQ / 16;
  int b = blockIdx.x;
  const int lt = b % LT; b /= LT;
  const int h  = b % HEADS; b /= HEADS;
  const int n  = b;
  const int l0 = lt * 16;
  const int lane = threadIdx.x & 31;
  const int ncol = lane & 15, hgrp = lane >> 4;
  const long tokbase = ((long)n * SEQ + l0) * EMBED + h * HDIM;
  const long nh = (long)n * HEADS + h;

  const float* srcs[3] = { query + tokbase, keys + tokbase, vals + tokbase };
  const __bf16* Ws[3]  = { Wqb, Wkb, Wvb };

#pragma unroll
  for (int p = 0; p < 3; ++p) {
    v16bf a0 = load_a_f32(srcs[p], EMBED);       // d = 0..31
    v16bf a1 = load_a_f32(srcs[p] + 32, EMBED);  // d = 32..63
#pragma unroll
    for (int t = 0; t < 4; ++t) {
      v8f c = {0.f, 0.f, 0.f, 0.f, 0.f, 0.f, 0.f, 0.f};
      c = wmma_bf16(a0, load_b_bf16(Ws[p] + t * 16 * HDIM,      HDIM), c);
      c = wmma_bf16(a1, load_b_bf16(Ws[p] + t * 16 * HDIM + 32, HDIM), c);
      if (p < 2) {
        __bf16* dst = (p == 0) ? Qp : Kp;   // [n,h,l,d] row-major
#pragma unroll
        for (int r = 0; r < 8; ++r)
          dst[(nh * SEQ + l0 + r + 8 * hgrp) * HDIM + t * 16 + ncol] = (__bf16)c[r];
      } else {                               // V transposed: [n,h,d,l]
#pragma unroll
        for (int r = 0; r < 8; ++r)
          Vt[(nh * HDIM + t * 16 + ncol) * SEQ + l0 + r + 8 * hgrp] = (__bf16)c[r];
      }
    }
  }
}

// ---------------- Kernel 3: flash attention (WMMA + async LDS staging) ----------------
// 256 threads = 8 waves share one (n,h); each wave owns 16 Q rows (128 per block).
// K/V tiles (32 keys) double-buffered in LDS via GLOBAL_LOAD_ASYNC_TO_LDS_B128.
__global__ void __launch_bounds__(256) attn_fwd(
    const __bf16* __restrict__ Qp, const __bf16* __restrict__ Kp,
    const __bf16* __restrict__ Vt, __bf16* __restrict__ X) {
  __shared__ __bf16 kbuf[2][32 * 72] __attribute__((aligned(16)));  // [key][d], +8 pad
  __shared__ __bf16 vbuf[2][64 * 40] __attribute__((aligned(16)));  // [d][l],  +8 pad
  __shared__ __bf16 pbuf[8][16 * 40] __attribute__((aligned(16)));  // per-wave P staging

  const int QB = SEQ / 128;
  int b = blockIdx.x;
  const int qb = b % QB; b /= QB;
  const int h  = b % HEADS; b /= HEADS;
  const int n  = b;
  const int w  = (int)threadIdx.x >> 5;
  const int q0 = qb * 128 + w * 16;
  const int lane = threadIdx.x & 31;
  const int ncol = lane & 15, hgrp = lane >> 4;
  const long nh = (long)n * HEADS + h;
  const float scale = 0.03125f;  // 1/sqrt(EMBED)

  const __bf16* Kbase = Kp + nh * SEQ * HDIM;
  const __bf16* Vbase = Vt + nh * HDIM * SEQ;

  auto issue_tiles = [&](int buf, int kt) {
    const int t = (int)threadIdx.x;
    {  // K tile: 32 rows x 128B = 256 x 16B chunks
      const int row = t >> 3, seg = t & 7;
      async_b128((unsigned)(uintptr_t)&kbuf[buf][row * 72 + seg * 8],
                 Kbase + (long)(kt + row) * HDIM + seg * 8);
    }
    {  // V tile: 64 d-rows x 64B = 256 x 16B chunks
      const int row = t >> 2, seg = t & 3;
      async_b128((unsigned)(uintptr_t)&vbuf[buf][row * 40 + seg * 8],
                 Vbase + (long)row * SEQ + kt + seg * 8);
    }
  };

  const v16bf aq0 = load_a_bf16(Qp + (nh * SEQ + q0) * HDIM,      HDIM);
  const v16bf aq1 = load_a_bf16(Qp + (nh * SEQ + q0) * HDIM + 32, HDIM);

  float mrow[8], lrow[8];
  v8f o[4];
#pragma unroll
  for (int r = 0; r < 8; ++r) { mrow[r] = -1e30f; lrow[r] = 0.f; }
#pragma unroll
  for (int t = 0; t < 4; ++t) o[t] = (v8f){0.f, 0.f, 0.f, 0.f, 0.f, 0.f, 0.f, 0.f};

  issue_tiles(0, 0);
  for (int kt = 0; kt < SEQ; kt += 32) {
    const int cur = (kt >> 5) & 1;
    asm volatile("s_wait_asynccnt 0x0" ::: "memory");  // own async copies landed
    __syncthreads();  // everyone's copies landed; prior iter's LDS reads retired
    if (kt + 32 < SEQ) issue_tiles(cur ^ 1, kt + 32);  // overlap next DMA with compute

    // scores: contract d=64, keys kt..kt+15 (s0) and kt+16..kt+31 (s1)
    v8f s0 = {0.f, 0.f, 0.f, 0.f, 0.f, 0.f, 0.f, 0.f};
    v8f s1 = {0.f, 0.f, 0.f, 0.f, 0.f, 0.f, 0.f, 0.f};
    s0 = wmma_bf16(aq0, load_b_bf16(&kbuf[cur][0],           72), s0);
    s0 = wmma_bf16(aq1, load_b_bf16(&kbuf[cur][32],          72), s0);
    s1 = wmma_bf16(aq0, load_b_bf16(&kbuf[cur][16 * 72],     72), s1);
    s1 = wmma_bf16(aq1, load_b_bf16(&kbuf[cur][16 * 72 + 32],72), s1);

#pragma unroll
    for (int r = 0; r < 8; ++r) {
      float x0 = s0[r] * scale, x1 = s1[r] * scale;
      float rm = fmaxf(x0, x1);
      rm = fmaxf(rm, __shfl_xor(rm, 1));
      rm = fmaxf(rm, __shfl_xor(rm, 2));
      rm = fmaxf(rm, __shfl_xor(rm, 4));
      rm = fmaxf(rm, __shfl_xor(rm, 8));
      float nm = fmaxf(mrow[r], rm);
      float alpha = __expf(mrow[r] - nm);
      float p0 = __expf(x0 - nm), p1 = __expf(x1 - nm);
      float rs = p0 + p1;
      rs += __shfl_xor(rs, 1);
      rs += __shfl_xor(rs, 2);
      rs += __shfl_xor(rs, 4);
      rs += __shfl_xor(rs, 8);
      lrow[r] = lrow[r] * alpha + rs;
      mrow[r] = nm;
      pbuf[w][(r + 8 * hgrp) * 40 + ncol]      = (__bf16)p0;
      pbuf[w][(r + 8 * hgrp) * 40 + 16 + ncol] = (__bf16)p1;
#pragma unroll
      for (int t = 0; t < 4; ++t) o[t][r] *= alpha;
    }
    __builtin_amdgcn_wave_barrier();               // order P store -> P load (same wave)
    const v16bf ap = load_a_bf16(pbuf[w], 40);     // P as 16x32 A fragment
#pragma unroll
    for (int t = 0; t < 4; ++t)
      o[t] = wmma_bf16(ap, load_b_bf16(&vbuf[cur][t * 16 * 40], 40), o[t]);
    __builtin_amdgcn_wave_barrier();               // WAR guard on pbuf
  }

#pragma unroll
  for (int r = 0; r < 8; ++r) {
    const float inv = 1.f / lrow[r];
    const long row = (long)n * SEQ + q0 + r + 8 * hgrp;
#pragma unroll
    for (int t = 0; t < 4; ++t)
      X[row * EMBED + h * HDIM + t * 16 + ncol] = (__bf16)(o[t][r] * inv);
  }
}

// ---------------- Kernel 4: output projection GEMM (WMMA) ----------------
// out[8192,1024] = X @ Wo^T + bo. Waves in a block share one 64-col Wo tile.
__global__ void __launch_bounds__(128) out_proj(
    const __bf16* __restrict__ X, const __bf16* __restrict__ Wob,
    const float* __restrict__ bo, float* __restrict__ out) {
  const int NT = EMBED / 64;                         // 16 column tiles
  const int ntile = blockIdx.x % NT;
  const int mt = (blockIdx.x / NT) * 4 + ((int)threadIdx.x >> 5);
  const long r0 = (long)mt * 16;
  const int o0 = ntile * 64;
  const int lane = threadIdx.x & 31;
  const int ncol = lane & 15, hgrp = lane >> 4;

  v8f c[4];
#pragma unroll
  for (int t = 0; t < 4; ++t) c[t] = (v8f){0.f, 0.f, 0.f, 0.f, 0.f, 0.f, 0.f, 0.f};

  for (int k = 0; k < EMBED; k += 32) {
    const v16bf a = load_a_bf16(X + r0 * EMBED + k, EMBED);
#pragma unroll
    for (int t = 0; t < 4; ++t)
      c[t] = wmma_bf16(a, load_b_bf16(Wob + (long)(o0 + t * 16) * EMBED + k, EMBED), c[t]);
  }
#pragma unroll
  for (int t = 0; t < 4; ++t) {
    const float bias = bo[o0 + t * 16 + ncol];
#pragma unroll
    for (int r = 0; r < 8; ++r)
      out[(r0 + r + 8 * hgrp) * EMBED + o0 + t * 16 + ncol] = c[t][r] + bias;
  }
}

extern "C" void kernel_launch(void* const* d_in, const int* in_sizes, int n_in,
                              void* d_out, int out_size, void* d_ws, size_t ws_size,
                              hipStream_t stream) {
  (void)in_sizes; (void)n_in; (void)out_size; (void)ws_size;
  const float* vals  = (const float*)d_in[0];
  const float* keys  = (const float*)d_in[1];
  const float* query = (const float*)d_in[2];
  const float* Wv = (const float*)d_in[3];
  const float* Wk = (const float*)d_in[4];
  const float* Wq = (const float*)d_in[5];
  const float* Wo = (const float*)d_in[6];
  const float* bo = (const float*)d_in[7];
  float* out = (float*)d_out;

  char* ws = (char*)d_ws;
  const size_t SZ = (size_t)BATCH * HEADS * SEQ * HDIM * sizeof(__bf16);  // 16 MB
  __bf16* Qp  = (__bf16*)(ws);
  __bf16* Kp  = (__bf16*)(ws + SZ);
  __bf16* Vt  = (__bf16*)(ws + 2 * SZ);
  __bf16* X   = (__bf16*)(ws + 3 * SZ);
  __bf16* Wqb = (__bf16*)(ws + 4 * SZ);
  __bf16* Wkb = Wqb + HDIM * HDIM;
  __bf16* Wvb = Wkb + HDIM * HDIM;
  __bf16* Wob = Wvb + HDIM * HDIM;

  cvt_weights<<<(EMBED * EMBED + 255) / 256, 256, 0, stream>>>(Wv, Wk, Wq, Wo, Wvb, Wkb, Wqb, Wob);
  proj_qkv<<<BATCH * HEADS * (SEQ / 16), 32, 0, stream>>>(query, keys, vals, Wqb, Wkb, Wvb, Qp, Kp, Vt);
  attn_fwd<<<BATCH * HEADS * (SEQ / 128), 256, 0, stream>>>(Qp, Kp, Vt, X);
  out_proj<<<(BATCH * SEQ / 64) * (EMBED / 64), 128, 0, stream>>>(X, Wob, bo, out);
}
